// PEER_41472204210327
// MI455X (gfx1250) — matrix operand order repaired and checked by reading, not compiled, not run on hardware
//
#include <hip/hip_runtime.h>
#include <hip/hip_bf16.h>

// ---------------------------------------------------------------------------
// PEER expert-retrieval layer for MI455X (gfx1250, wave32).
// B=2, T=1024, D=1024, H=8, DK=128, K=8, NKEYS=256  -> 2048 tokens.
//
// Roofline: expert gather stage moves ~1.07 GB of random 4KB rows from
// 512 MB tables (> 192 MB L2) -> HBM bound (~46 us at 23.3 TB/s). GEMMs
// (~11 GF) are trivial on the bf16 WMMA path. So: WMMA for qproj/sims,
// TDM gather-mode DMA (tensor_load_to_lds) for the expert row streaming,
// double-buffered in LDS so waves compute while the next chunk is in flight.
// ---------------------------------------------------------------------------

typedef __attribute__((ext_vector_type(16))) __bf16       v16bf;
typedef __attribute__((ext_vector_type(8)))  float        v8f;
typedef __attribute__((ext_vector_type(4)))  float        f32x4;
typedef __attribute__((ext_vector_type(4)))  unsigned int u32x4;
typedef __attribute__((ext_vector_type(8)))  int          i32x8;
typedef __attribute__((ext_vector_type(4)))  int          i32x4;

union ABFrag { v16bf v; u32x4 u[2]; };

#define NTOK   2048
#define DIM    1024
#define NHEAD  8
#define DKEY   128
#define TOPK   8
#define NKEY   256

// ---- workspace layout (bytes) ----
#define OFF_XN    (size_t)0                                   // f32  [2048,1024]
#define OFF_XNB   (OFF_XN   + (size_t)NTOK*DIM*4)             // bf16 [2048,1024]
#define OFF_WQB   (OFF_XNB  + (size_t)NTOK*DIM*2)             // bf16 [2048,1024]
#define OFF_KEYB  (OFF_WQB  + (size_t)2048*1024*2)            // bf16 [8,256,2,128]
#define OFF_QB    (OFF_KEYB + (size_t)8*256*2*128*2)          // bf16 [2048,2048]
#define OFF_SIM   (OFF_QB   + (size_t)NTOK*2048*2)            // f32  [2048,8,2,256]
#define OFF_FI    (OFF_SIM  + (size_t)NTOK*NHEAD*2*NKEY*4)    // i32  [2048,8,8]
#define OFF_FS    (OFF_FI   + (size_t)NTOK*NHEAD*TOPK*4)      // f32  [2048,8,8]

__device__ __forceinline__ float gelu_exact(float x) {
    return 0.5f * x * (1.0f + erff(x * 0.70710678118654752f));
}

// ---------------------------------------------------------------------------
// f32 -> bf16 bulk convert (w_q, keys)
// ---------------------------------------------------------------------------
__global__ void peer_cvt_bf16(const float* __restrict__ in, __bf16* __restrict__ out, int n) {
    int i = blockIdx.x * blockDim.x + threadIdx.x;
    int stride = gridDim.x * blockDim.x;
    for (; i < n; i += stride) out[i] = (__bf16)in[i];
}

// ---------------------------------------------------------------------------
// RMSNorm: one 256-thread block per token. Writes f32 xn and bf16 copy.
// ---------------------------------------------------------------------------
__global__ void peer_rmsnorm(const float* __restrict__ x, const float* __restrict__ nw,
                             float* __restrict__ xn, __bf16* __restrict__ xnb) {
    const int token = blockIdx.x;
    const int tid   = threadIdx.x;
    const int lane  = tid & 31, w = tid >> 5;
    __shared__ float red[8];

    f32x4 xv = *(const f32x4*)(x + (size_t)token * DIM + tid * 4);
    float ss = xv[0]*xv[0] + xv[1]*xv[1] + xv[2]*xv[2] + xv[3]*xv[3];
    #pragma unroll
    for (int off = 16; off; off >>= 1) ss += __shfl_xor(ss, off, 32);
    if (lane == 0) red[w] = ss;
    __syncthreads();
    float tot = red[0]+red[1]+red[2]+red[3]+red[4]+red[5]+red[6]+red[7];
    float inv = rsqrtf(tot * (1.0f / (float)DIM) + 1e-6f);

    f32x4 wv = *(const f32x4*)(nw + tid * 4);
    f32x4 y;
    y[0] = xv[0]*inv*wv[0]; y[1] = xv[1]*inv*wv[1];
    y[2] = xv[2]*inv*wv[2]; y[3] = xv[3]*inv*wv[3];
    *(f32x4*)(xn + (size_t)token * DIM + tid * 4) = y;
    __bf16* ob = xnb + (size_t)token * DIM + tid * 4;
    ob[0] = (__bf16)y[0]; ob[1] = (__bf16)y[1]; ob[2] = (__bf16)y[2]; ob[3] = (__bf16)y[3];
}

// ---------------------------------------------------------------------------
// Q projection: q[2048,2048] = xnb[2048,1024] @ wqb^T   (wqb stored [N,K])
// Block = 256 thr (8 waves). Tile: M=128 (16 rows/wave), N=64, K-step 32.
// ---------------------------------------------------------------------------
__global__ void peer_qproj(const __bf16* __restrict__ xnb, const __bf16* __restrict__ wqb,
                           __bf16* __restrict__ qb) {
    const int lane = threadIdx.x & 31;
    const int w    = threadIdx.x >> 5;
    const int l16  = lane & 15;
    const int half = lane >> 4;
    const int m0   = blockIdx.x * 128 + w * 16;
    const int n0   = blockIdx.y * 64;

    v8f acc[4] = {};
    for (int k0 = 0; k0 < DIM; k0 += 32) {
        ABFrag a;
        const __bf16* ap = xnb + (size_t)(m0 + l16) * DIM + k0 + half * 8;
        a.u[0] = *(const u32x4*)(ap);
        a.u[1] = *(const u32x4*)(ap + 16);
        #pragma unroll
        for (int nb = 0; nb < 4; ++nb) {
            ABFrag b;
            const __bf16* bp = wqb + (size_t)(n0 + nb * 16 + l16) * DIM + k0 + half * 8;
            b.u[0] = *(const u32x4*)(bp);
            b.u[1] = *(const u32x4*)(bp + 16);
            acc[nb] = __builtin_amdgcn_wmma_f32_16x16x32_bf16(
                false, a.v, false, b.v, (short)0, acc[nb], false, false);
        }
    }
    #pragma unroll
    for (int nb = 0; nb < 4; ++nb)
        #pragma unroll
        for (int r = 0; r < 8; ++r) {
            int row = m0 + half * 8 + r;
            int col = n0 + nb * 16 + l16;
            qb[(size_t)row * 2048 + col] = (__bf16)acc[nb][r];
        }
}

// ---------------------------------------------------------------------------
// Similarities: for (head h, pair p): sim = q[:,h,p,:] @ keys[h,:,p,:]^T
// ---------------------------------------------------------------------------
__global__ void peer_sim(const __bf16* __restrict__ qb, const __bf16* __restrict__ keysb,
                         float* __restrict__ sim) {
    const int lane = threadIdx.x & 31;
    const int w    = threadIdx.x >> 5;
    const int l16  = lane & 15;
    const int half = lane >> 4;
    const int t0   = blockIdx.x * 16;
    const int h    = blockIdx.y >> 1;
    const int p    = blockIdx.y & 1;

    v8f acc[2] = {};
    for (int k0 = 0; k0 < DKEY; k0 += 32) {
        ABFrag a;
        const __bf16* ap = qb + (size_t)(t0 + l16) * 2048 + h * 256 + p * 128 + k0 + half * 8;
        a.u[0] = *(const u32x4*)(ap);
        a.u[1] = *(const u32x4*)(ap + 16);
        #pragma unroll
        for (int nb = 0; nb < 2; ++nb) {
            int n = w * 32 + nb * 16 + l16;
            ABFrag b;
            const __bf16* bp = keysb + ((size_t)(h * NKEY + n) * 2 + p) * DKEY + k0 + half * 8;
            b.u[0] = *(const u32x4*)(bp);
            b.u[1] = *(const u32x4*)(bp + 16);
            acc[nb] = __builtin_amdgcn_wmma_f32_16x16x32_bf16(
                false, a.v, false, b.v, (short)0, acc[nb], false, false);
        }
    }
    #pragma unroll
    for (int nb = 0; nb < 2; ++nb)
        #pragma unroll
        for (int r = 0; r < 8; ++r) {
            int tok = t0 + half * 8 + r;
            int key = w * 32 + nb * 16 + l16;
            sim[(((size_t)tok * NHEAD + h) * 2 + p) * NKEY + key] = acc[nb][r];
        }
}

// ---------------------------------------------------------------------------
// Top-k: one wave per (token, head); wave32 shfl_xor argmax rounds.
// ---------------------------------------------------------------------------
__global__ void peer_topk(const float* __restrict__ sim, int* __restrict__ fi,
                          float* __restrict__ fs) {
    const int lane = threadIdx.x & 31;
    const int w    = threadIdx.x >> 5;
    const int pair = blockIdx.x * 8 + w;          // token*8 + h
    const float NEG = -3.0e38f;

    const float* s1p = sim + ((size_t)pair * 2 + 0) * NKEY;
    const float* s2p = sim + ((size_t)pair * 2 + 1) * NKEY;
    float v1[8], v2[8];
    #pragma unroll
    for (int j = 0; j < 8; ++j) { v1[j] = s1p[lane + 32 * j]; v2[j] = s2p[lane + 32 * j]; }

    float s1v[8], s2v[8];
    int   i1v[8], i2v[8];
    #pragma unroll
    for (int r = 0; r < 8; ++r) {
        float best = NEG; int bidx = lane;
        #pragma unroll
        for (int j = 0; j < 8; ++j) { int idx = lane + 32 * j; if (v1[j] > best) { best = v1[j]; bidx = idx; } }
        #pragma unroll
        for (int off = 16; off; off >>= 1) {
            float ov = __shfl_xor(best, off, 32);
            int   oi = __shfl_xor(bidx, off, 32);
            if (ov > best || (ov == best && oi < bidx)) { best = ov; bidx = oi; }
        }
        s1v[r] = best; i1v[r] = bidx;
        if ((bidx & 31) == lane) v1[bidx >> 5] = NEG;
    }
    #pragma unroll
    for (int r = 0; r < 8; ++r) {
        float best = NEG; int bidx = lane;
        #pragma unroll
        for (int j = 0; j < 8; ++j) { int idx = lane + 32 * j; if (v2[j] > best) { best = v2[j]; bidx = idx; } }
        #pragma unroll
        for (int off = 16; off; off >>= 1) {
            float ov = __shfl_xor(best, off, 32);
            int   oi = __shfl_xor(bidx, off, 32);
            if (ov > best || (ov == best && oi < bidx)) { best = ov; bidx = oi; }
        }
        s2v[r] = best; i2v[r] = bidx;
        if ((bidx & 31) == lane) v2[bidx >> 5] = NEG;
    }

    float cs[2];
    #pragma unroll
    for (int t = 0; t < 2; ++t) { int c = lane + 32 * t; cs[t] = s1v[c >> 3] + s2v[c & 7]; }
    #pragma unroll
    for (int r = 0; r < 8; ++r) {
        float best = NEG; int bc = lane;
        #pragma unroll
        for (int t = 0; t < 2; ++t) { int c = lane + 32 * t; if (cs[t] > best) { best = cs[t]; bc = c; } }
        #pragma unroll
        for (int off = 16; off; off >>= 1) {
            float ov = __shfl_xor(best, off, 32);
            int   oc = __shfl_xor(bc, off, 32);
            if (ov > best || (ov == best && oc < bc)) { best = ov; bc = oc; }
        }
        if ((bc & 31) == lane) cs[bc >> 5] = NEG;
        if (lane == 0) {
            fi[(size_t)pair * TOPK + r] = i1v[bc >> 3] * NKEY + i2v[bc & 7];
            fs[(size_t)pair * TOPK + r] = best;
        }
    }
}

// ---------------------------------------------------------------------------
// TDM gather-mode descriptor: pull EXP_CHUNK indexed rows (each DIM f32,
// 4KB) from a [65536, 1024] f32 table into LDS at lds_off.
//   group0: count=1 | gather_mode | 16-bit idx; lds_addr; global_addr; type=2
//   group1: data_size=4B; tensor_dim0=1024; tensor_dim1=65536;
//           tile_dim0=1024; tile_dim1=#indices; tensor_dim0_stride=1024
//   group2: packed 16-bit row indices
// This toolchain's builtin takes 6 args (extra int32x8 beyond the four ISA
// D# groups) -> zero-fill it.
// ---------------------------------------------------------------------------
#define EXP_CHUNK  4
#define EXP_NCHUNK 16

__device__ __forceinline__ void tdm_gather4(const float* __restrict__ base,
                                            const int* __restrict__ s_idx,
                                            unsigned lds_off) {
    unsigned long long ga = (unsigned long long)(size_t)base;
    u32x4 g0;
    g0[0] = 0x80000001u;                                   // count=1, gather_mode=1, idx16
    g0[1] = __builtin_amdgcn_readfirstlane(lds_off);       // lds_addr
    g0[2] = (unsigned)(ga & 0xFFFFFFFFu);                  // global_addr[31:0]
    g0[3] = (unsigned)((ga >> 32) & 0x01FFFFFFu) | 0x80000000u;  // addr[56:32] | type=2

    i32x8 g1;
    g1[0] = 0x00020000;                 // data_size=2 (4 bytes)
    g1[1] = (int)(((unsigned)DIM & 0xFFFFu) << 16);   // tensor_dim0 low16 @ bits 63:48
    g1[2] = 0;                          // tensor_dim0 hi16 = 0, tensor_dim1 low16 = 0
    g1[3] = (int)(1u | ((unsigned)DIM << 16));        // tensor_dim1 hi16 (65536), tile_dim0
    g1[4] = EXP_CHUNK;                  // tile_dim1 = #valid indices
    g1[5] = DIM;                        // tensor_dim0_stride low32
    g1[6] = 0;
    g1[7] = 0;

    unsigned p0 = ((unsigned)s_idx[0] & 0xFFFFu) | (((unsigned)s_idx[1] & 0xFFFFu) << 16);
    unsigned p1 = ((unsigned)s_idx[2] & 0xFFFFu) | (((unsigned)s_idx[3] & 0xFFFFu) << 16);
    i32x4 g2;
    g2[0] = (int)__builtin_amdgcn_readfirstlane(p0);
    g2[1] = (int)__builtin_amdgcn_readfirstlane(p1);
    g2[2] = 0; g2[3] = 0;
    i32x4 g3 = (i32x4)0;
    i32x8 gx = (i32x8)0;                // extra arg on this toolchain: zero-fill

    __builtin_amdgcn_tensor_load_to_lds(g0, g1, g2, g3, gx, 0);
}

// ---------------------------------------------------------------------------
// Expert gather + rank-1 down/up: one 256-thread block per token.
// Double-buffered TDM gather (4 rows / 16KB per op); waves compute from LDS
// while the next chunk streams in. Two barriers per 4 experts instead of two
// per expert; dot reductions stay in-wave (shfl_xor), partials parked in LDS.
// ---------------------------------------------------------------------------
__global__ void peer_experts(const float* __restrict__ xn, const float* __restrict__ down,
                             const float* __restrict__ up, const int* __restrict__ fi,
                             const float* __restrict__ fs, float* __restrict__ out) {
    __shared__ float s_buf[2][EXP_CHUNK * DIM];   // 2 x 16 KB
    __shared__ int   s_exp[64];
    __shared__ float s_sc[64];
    __shared__ float s_red[64 * 8];
    __shared__ float s_h[64];

    const int token = blockIdx.x;
    const int tid   = threadIdx.x;
    const int lane  = tid & 31, w = tid >> 5;

    if (tid < 64) {
        s_exp[tid] = fi[(size_t)token * 64 + tid];
        s_sc[tid]  = fs[(size_t)token * 64 + tid];
    }
    __syncthreads();

    f32x4 xv = *(const f32x4*)(xn + (size_t)token * DIM + tid * 4);
    const unsigned lds0 = (unsigned)(size_t)&s_buf[0][0];
    const unsigned lds1 = (unsigned)(size_t)&s_buf[1][0];

    // ---------------- Phase A: down rows -> hidden dots ----------------
    if (w == 0) tdm_gather4(down, &s_exp[0], lds0);
    for (int c = 0; c < EXP_NCHUNK; ++c) {
        if (w == 0) {
            if (c + 1 < EXP_NCHUNK) {
                tdm_gather4(down, &s_exp[(c + 1) * EXP_CHUNK], ((c + 1) & 1) ? lds1 : lds0);
                __builtin_amdgcn_s_wait_tensorcnt((short)1);   // chunk c landed
            } else {
                __builtin_amdgcn_s_wait_tensorcnt((short)0);
            }
        }
        __syncthreads();
        const float* buf = (c & 1) ? &s_buf[1][0] : &s_buf[0][0];
        #pragma unroll
        for (int r = 0; r < EXP_CHUNK; ++r) {
            f32x4 wd = *(const f32x4*)(buf + r * DIM + tid * 4);
            float part = wd[0]*xv[0] + wd[1]*xv[1] + wd[2]*xv[2] + wd[3]*xv[3];
            #pragma unroll
            for (int off = 16; off; off >>= 1) part += __shfl_xor(part, off, 32);
            if (lane == 0) s_red[(c * EXP_CHUNK + r) * 8 + w] = part;
        }
        __syncthreads();
    }

    if (tid < 64) {
        float d = 0.0f;
        #pragma unroll
        for (int j = 0; j < 8; ++j) d += s_red[tid * 8 + j];
        s_h[tid] = gelu_exact(d) * fmaxf(s_sc[tid], 0.0f);
    }

    // ---------------- Phase B: up rows -> output accumulation ----------------
    if (w == 0) tdm_gather4(up, &s_exp[0], lds0);
    __syncthreads();                                   // s_h visible to all
    f32x4 acc = {};
    for (int c = 0; c < EXP_NCHUNK; ++c) {
        if (w == 0) {
            if (c + 1 < EXP_NCHUNK) {
                tdm_gather4(up, &s_exp[(c + 1) * EXP_CHUNK], ((c + 1) & 1) ? lds1 : lds0);
                __builtin_amdgcn_s_wait_tensorcnt((short)1);
            } else {
                __builtin_amdgcn_s_wait_tensorcnt((short)0);
            }
        }
        __syncthreads();
        const float* buf = (c & 1) ? &s_buf[1][0] : &s_buf[0][0];
        #pragma unroll
        for (int r = 0; r < EXP_CHUNK; ++r) {
            f32x4 wu = *(const f32x4*)(buf + r * DIM + tid * 4);
            float h = s_h[c * EXP_CHUNK + r];
            acc[0] += h * wu[0]; acc[1] += h * wu[1];
            acc[2] += h * wu[2]; acc[3] += h * wu[3];
        }
        __syncthreads();
    }
    *(f32x4*)(out + (size_t)token * DIM + tid * 4) = acc;
}

// ---------------------------------------------------------------------------
extern "C" void kernel_launch(void* const* d_in, const int* in_sizes, int n_in,
                              void* d_out, int out_size, void* d_ws, size_t ws_size,
                              hipStream_t stream) {
    const float* x      = (const float*)d_in[0];   // [2,1024,1024]
    const float* w_q    = (const float*)d_in[1];   // [2048,1024]
    const float* keys   = (const float*)d_in[2];   // [8,256,2,128]
    const float* e_down = (const float*)d_in[3];   // [65536,1024]
    const float* e_up   = (const float*)d_in[4];   // [65536,1024]
    const float* norm_w = (const float*)d_in[5];   // [1024]
    float* out = (float*)d_out;

    char* ws = (char*)d_ws;
    float*  xn    = (float*)(ws + OFF_XN);
    __bf16* xnb   = (__bf16*)(ws + OFF_XNB);
    __bf16* wqb   = (__bf16*)(ws + OFF_WQB);
    __bf16* keysb = (__bf16*)(ws + OFF_KEYB);
    __bf16* qb    = (__bf16*)(ws + OFF_QB);
    float*  sim   = (float*)(ws + OFF_SIM);
    int*    fi    = (int*)(ws + OFF_FI);
    float*  fs    = (float*)(ws + OFF_FS);

    peer_cvt_bf16<<<2048, 256, 0, stream>>>(w_q, wqb, 2048 * 1024);
    peer_cvt_bf16<<<512, 256, 0, stream>>>(keys, keysb, 8 * 256 * 2 * 128);

    peer_rmsnorm<<<NTOK, 256, 0, stream>>>(x, norm_w, xn, xnb);

    peer_qproj<<<dim3(NTOK / 128, 2048 / 64), 256, 0, stream>>>(xnb, wqb, qb);

    peer_sim<<<dim3(NTOK / 16, NHEAD * 2), 256, 0, stream>>>(qb, keysb, sim);

    peer_topk<<<NTOK, 256, 0, stream>>>(sim, fi, fs);

    peer_experts<<<NTOK, 256, 0, stream>>>(xn, e_down, e_up, fi, fs, out);
}
